// HadamardMoE_6640019440354
// MI455X (gfx1250) — matrix operand order repaired and checked by reading, not compiled, not run on hardware
//
#include <hip/hip_runtime.h>
#include <hip/hip_bf16.h>
#include <math.h>

// ---------------------------------------------------------------------------
// HadamardMoE (BitNet-ternary block-diagonal MoE) for gfx1250 / MI455X.
//
// Roofline: ~130MB HBM traffic (5.6us @ 23.3TB/s) vs ~22 GFLOP of matmul ->
// matrix-core bound. All matmuls run on v_wmma_f32_16x16x32_f16 (K=32 matches
// the 32-dim expert blocks exactly). Ternary weights {-g,0,+g} are baked into
// f16 and pre-packed into the exact WMMA B-operand lane layout so the hot
// loops load operands with single contiguous 32B reads per lane.
// ---------------------------------------------------------------------------

typedef __attribute__((ext_vector_type(16))) _Float16 v16h;
typedef __attribute__((ext_vector_type(8)))  _Float16 v8h;
typedef __attribute__((ext_vector_type(8)))  float    v8f;

#define N_EMBD   1024
#define N_EXP    32
#define BLK      32      // per-expert block dim
#define HID      128     // expert hidden
#define RHID     256     // router hidden
#define TOKENS   16384   // B*T
#define TTILES   1024    // TOKENS/16

// ---- workspace layout (bytes) ----
#define WS_GAMMA     0
#define WS_PG        256                       // gate packed: 32*8*512 f16 = 256KB
#define WS_PU        (WS_PG + 262144)
#define WS_PD        (WS_PU + 262144)
#define WS_P1        (WS_PD + 262144)          // W1 packed: 16*32*512 f16 = 512KB
#define WS_P2        (WS_P1 + 524288)          // W2 packed: 2*8*512 f16 = 16KB
#define WS_WTS       (WS_P2 + 16384)           // router weights: 16384*32 f32 = 2MB
#define WS_NEEDED    (WS_WTS + (size_t)TOKENS * N_EXP * 4)

// --------------------------------------------------------------------------
// 1) absmean gamma for each of the three expert weight tensors (131072 elems)
// --------------------------------------------------------------------------
__global__ __launch_bounds__(256) void absmean_k(const float* gate, const float* up,
                                                 const float* down, float* gamma) {
  const float* w = (blockIdx.x == 0) ? gate : (blockIdx.x == 1) ? up : down;
  __shared__ float red[256];
  float s = 0.0f;
  for (int i = threadIdx.x; i < 131072; i += 256) s += fabsf(w[i]);
  red[threadIdx.x] = s;
  __syncthreads();
  for (int off = 128; off > 0; off >>= 1) {
    if (threadIdx.x < (unsigned)off) red[threadIdx.x] += red[threadIdx.x + off];
    __syncthreads();
  }
  if (threadIdx.x == 0) gamma[blockIdx.x] = red[0] / 131072.0f + 1e-8f;
}

// --------------------------------------------------------------------------
// 2) ternary-quantize expert weights and pack into WMMA B-operand layout.
//    B tile (32x16): packed[tile*512 + lane*16 + j],
//      n = lane&15, k = (lane>>4)*8 + j + (j&8).
//    gate/up tiles: tile = e*8 + ht  (N = hidden h, K = d)
//    down tiles:    tile = e*8 + nt*4 + kc  (N = d, K = h, k = kc*32 + ...)
// --------------------------------------------------------------------------
__global__ __launch_bounds__(256) void pack_expert_k(const float* gate, const float* up,
                                                     const float* down, const float* gamma,
                                                     _Float16* pg, _Float16* pu, _Float16* pd) {
  int idx = blockIdx.x * 256 + threadIdx.x;   // 0 .. 3*131072-1
  int which = idx >> 17;
  int t = idx & 131071;
  int tile = t >> 9, r = t & 511, lane = r >> 4, j = r & 15;
  int e = tile >> 3;
  int n = lane & 15, hh = lane >> 4;
  float g, w;
  _Float16* dst;
  if (which < 2) {
    int ht = tile & 7;
    int h = ht * 16 + n;              // N dim: hidden
    int k = hh * 8 + j + (j & 8);     // K dim: d in 0..31
    const float* src = (which == 0) ? gate : up;
    g = gamma[which];
    w = src[(e * HID + h) * BLK + k];
    dst = (which == 0) ? pg : pu;
  } else {
    int sub = tile & 7, nt = sub >> 2, kc = sub & 3;
    int d = nt * 16 + n;              // N dim: output d
    int k = kc * 32 + hh * 8 + j + (j & 8);    // K dim: hidden h in 0..127
    g = gamma[2];
    w = down[(e * BLK + d) * HID + k];
    dst = pd;
  }
  float q = rintf(w / g);
  q = fminf(1.0f, fmaxf(-1.0f, q));
  dst[t] = (_Float16)(q * g);
}

// --------------------------------------------------------------------------
// 3) pack router weights (no quantization) into WMMA B layout, f32 -> f16.
//    W1 [256,1024]: tiles tl = nt*32 + kc (nt 0..15 over hidden, kc 0..31 over embd)
//    W2 [32,256]:   tiles tl = nt*8 + kc (nt 0..1 over experts, kc 0..7 over hidden)
// --------------------------------------------------------------------------
__global__ __launch_bounds__(256) void pack_router_k(const float* w1, const float* w2,
                                                     _Float16* p1, _Float16* p2) {
  int idx = blockIdx.x * 256 + threadIdx.x;   // 0 .. 270335
  if (idx < 262144) {
    int tile = idx >> 9, r = idx & 511, lane = r >> 4, j = r & 15;
    int nt = tile >> 5, kc = tile & 31;
    int n = nt * 16 + (lane & 15);            // hidden h
    int k = kc * 32 + (lane >> 4) * 8 + j + (j & 8);   // embd c
    p1[idx] = (_Float16)w1[n * N_EMBD + k];
  } else {
    int t = idx - 262144;                     // 0 .. 8191
    int tile = t >> 9, r = t & 511, lane = r >> 4, j = r & 15;
    int nt = tile >> 3, kc = tile & 7;
    int n = nt * 16 + (lane & 15);            // expert e
    int k = kc * 32 + (lane >> 4) * 8 + j + (j & 8);   // hidden h
    p2[t] = (_Float16)w2[n * RHID + k];
  }
}

// helper: build a 16x32 A operand from an LDS row base (two 16B loads)
__device__ __forceinline__ v16h load_a(const _Float16* row) {
  v8h alo = *(const v8h*)row;        // k = half*8 + 0..7
  v8h ahi = *(const v8h*)(row + 16); // k = 16 + half*8 + 0..7
  v16h a;
#pragma unroll
  for (int i = 0; i < 8; i++) { a[i] = alo[i]; a[8 + i] = ahi[i]; }
  return a;
}

// --------------------------------------------------------------------------
// 4) Router: per 16-token tile. 8 waves.
//    h = silu(x @ W1^T) via 16 n-tiles x 32 k-steps of WMMA;
//    logits = h @ W2^T via 2 n-tiles x 8 k-steps; softmax/top-4/renorm scalar.
// --------------------------------------------------------------------------
__global__ __launch_bounds__(256) void router_k(const float* __restrict__ x,
                                                const _Float16* __restrict__ p1,
                                                const _Float16* __restrict__ p2,
                                                float* __restrict__ weights) {
  __shared__ _Float16 xt[16 * N_EMBD];   // 32KB token tile, f16
  __shared__ _Float16 h1[16 * RHID];     // 8KB hidden, f16
  __shared__ float    lg[16 * N_EXP];    // 2KB logits
  const int tid = threadIdx.x;
  const int t0 = blockIdx.x * 16;

  // stage x tile as f16 (float4-vectorized global reads)
  const float4* xs = (const float4*)(x + (size_t)t0 * N_EMBD);
  for (int q = tid; q < 4096; q += 256) {
    float4 v = xs[q];
    int b = q * 4;
    xt[b + 0] = (_Float16)v.x; xt[b + 1] = (_Float16)v.y;
    xt[b + 2] = (_Float16)v.z; xt[b + 3] = (_Float16)v.w;
  }
  __syncthreads();

  const int lane = tid & 31, wave = tid >> 5;
  const int mrow = lane & 15, half = lane >> 4;

  // ---- matmul 1 + SiLU ----
  for (int nt = wave * 2; nt < wave * 2 + 2; nt++) {
    v8f acc = {};
    for (int kc = 0; kc < 32; kc++) {
      v16h a = load_a(&xt[mrow * N_EMBD + kc * 32 + half * 8]);
      v16h b = *(const v16h*)&p1[(size_t)(nt * 32 + kc) * 512 + lane * 16];
      acc = __builtin_amdgcn_wmma_f32_16x16x32_f16(false, a, false, b, (short)0, acc, false, false);
    }
#pragma unroll
    for (int v = 0; v < 8; v++) {
      float g = acc[v];
      float s = g / (1.0f + __expf(-g));           // silu
      int row = v + half * 8;
      h1[row * RHID + nt * 16 + mrow] = (_Float16)s;
    }
  }
  __syncthreads();

  // ---- matmul 2: logits (waves 0,1) ----
  if (wave < 2) {
    int nt = wave;
    v8f acc = {};
    for (int kc = 0; kc < 8; kc++) {
      v16h a = load_a(&h1[mrow * RHID + kc * 32 + half * 8]);
      v16h b = *(const v16h*)&p2[(size_t)(nt * 8 + kc) * 512 + lane * 16];
      acc = __builtin_amdgcn_wmma_f32_16x16x32_f16(false, a, false, b, (short)0, acc, false, false);
    }
#pragma unroll
    for (int v = 0; v < 8; v++) {
      int row = v + half * 8;
      lg[row * N_EXP + nt * 16 + mrow] = acc[v];
    }
  }
  __syncthreads();

  // ---- softmax + top-4 mask + renormalize (one thread per token) ----
  if (tid < 16) {
    float p[N_EXP];
    float mx = -1e30f;
    for (int e = 0; e < N_EXP; e++) { p[e] = lg[tid * N_EXP + e]; mx = fmaxf(mx, p[e]); }
    float s = 0.0f;
    for (int e = 0; e < N_EXP; e++) { p[e] = __expf(p[e] - mx); s += p[e]; }
    float inv = 1.0f / s;
    for (int e = 0; e < N_EXP; e++) p[e] *= inv;
    unsigned sel = 0;
    for (int r = 0; r < 4; r++) {        // 4 max-passes == jax top_k tie-break
      float best = -1.0f; int bi = 0;
      for (int e = 0; e < N_EXP; e++)
        if (!((sel >> e) & 1u) && p[e] > best) { best = p[e]; bi = e; }
      sel |= 1u << bi;
    }
    float msum = 0.0f;
    for (int e = 0; e < N_EXP; e++) if ((sel >> e) & 1u) msum += p[e];
    float invm = 1.0f / fmaxf(msum, 1e-8f);
    for (int e = 0; e < N_EXP; e++)
      weights[(size_t)(t0 + tid) * N_EXP + e] = ((sel >> e) & 1u) ? p[e] * invm : 0.0f;
  }
}

// --------------------------------------------------------------------------
// 5) Experts: per 16-token tile, 8 waves x 4 iterations cover 32 experts.
//    Per (wave, expert): A = x block 16x32; 8 h-tiles of gate+up WMMA;
//    SwiGLU -> per-wave LDS (D-layout -> A-layout transpose); down = 2 n-tiles
//    x 4 k-steps WMMA; scale by router weight; store f32 output.
//    24 v_wmma per wave-expert, 768 per block.
// --------------------------------------------------------------------------
__global__ __launch_bounds__(256) void expert_k(const float* __restrict__ x,
                                                const _Float16* __restrict__ pg,
                                                const _Float16* __restrict__ pu,
                                                const _Float16* __restrict__ pd,
                                                const float* __restrict__ weights,
                                                float* __restrict__ out) {
  __shared__ _Float16 xt[16 * N_EMBD];     // 32KB token tile
  __shared__ _Float16 hidb[8 * 16 * HID];  // 32KB per-wave SwiGLU staging
  const int tid = threadIdx.x;
  const int t0 = blockIdx.x * 16;

  const float4* xs = (const float4*)(x + (size_t)t0 * N_EMBD);
  for (int q = tid; q < 4096; q += 256) {
    float4 v = xs[q];
    int b = q * 4;
    xt[b + 0] = (_Float16)v.x; xt[b + 1] = (_Float16)v.y;
    xt[b + 2] = (_Float16)v.z; xt[b + 3] = (_Float16)v.w;
  }
  __syncthreads();

  const int lane = tid & 31, wave = tid >> 5;
  const int mrow = lane & 15, half = lane >> 4;
  _Float16* hw = &hidb[wave * 16 * HID];   // private to this wave; DS in-order

  for (int ei = 0; ei < 4; ei++) {
    const int e = ei * 8 + wave;

    // A = this expert's 32-dim x block (16x32)
    v16h a = load_a(&xt[mrow * N_EMBD + e * BLK + half * 8]);

    // gate & up, SwiGLU -> LDS in A-consumable layout
    for (int ht = 0; ht < 8; ht++) {
      v16h bg = *(const v16h*)&pg[(size_t)(e * 8 + ht) * 512 + lane * 16];
      v16h bu = *(const v16h*)&pu[(size_t)(e * 8 + ht) * 512 + lane * 16];
      v8f cg = {}, cu = {};
      cg = __builtin_amdgcn_wmma_f32_16x16x32_f16(false, a, false, bg, (short)0, cg, false, false);
      cu = __builtin_amdgcn_wmma_f32_16x16x32_f16(false, a, false, bu, (short)0, cu, false, false);
#pragma unroll
      for (int v = 0; v < 8; v++) {
        float g = cg[v];
        float h = (g / (1.0f + __expf(-g))) * cu[v];   // silu(g)*u
        int row = v + half * 8;
        hw[row * HID + ht * 16 + mrow] = (_Float16)h;
      }
    }

    // router weights for the 8 output rows this lane owns
    float wreg[8];
#pragma unroll
    for (int v = 0; v < 8; v++)
      wreg[v] = weights[(size_t)(t0 + v + half * 8) * N_EXP + e];

    // down projection: D=32 -> 2 n-tiles; K=128 -> 4 k-steps
    for (int nt = 0; nt < 2; nt++) {
      v8f acc = {};
      for (int kc = 0; kc < 4; kc++) {
        v16h ah = load_a(&hw[mrow * HID + kc * 32 + half * 8]);
        v16h b = *(const v16h*)&pd[(size_t)(e * 8 + nt * 4 + kc) * 512 + lane * 16];
        acc = __builtin_amdgcn_wmma_f32_16x16x32_f16(false, ah, false, b, (short)0, acc, false, false);
      }
#pragma unroll
      for (int v = 0; v < 8; v++) {
        int row = v + half * 8;
        out[(size_t)(t0 + row) * N_EMBD + e * BLK + nt * 16 + mrow] = acc[v] * wreg[v];
      }
    }
  }
}

// --------------------------------------------------------------------------
extern "C" void kernel_launch(void* const* d_in, const int* in_sizes, int n_in,
                              void* d_out, int out_size, void* d_ws, size_t ws_size,
                              hipStream_t stream) {
  (void)in_sizes; (void)n_in; (void)out_size;
  if (ws_size < WS_NEEDED) return;

  const float* x  = (const float*)d_in[0];
  const float* w1 = (const float*)d_in[1];
  const float* w2 = (const float*)d_in[2];
  const float* gw = (const float*)d_in[3];
  const float* uw = (const float*)d_in[4];
  const float* dw = (const float*)d_in[5];
  float* out = (float*)d_out;

  char* ws = (char*)d_ws;
  float*    gamma = (float*)(ws + WS_GAMMA);
  _Float16* pg    = (_Float16*)(ws + WS_PG);
  _Float16* pu    = (_Float16*)(ws + WS_PU);
  _Float16* pd    = (_Float16*)(ws + WS_PD);
  _Float16* p1    = (_Float16*)(ws + WS_P1);
  _Float16* p2    = (_Float16*)(ws + WS_P2);
  float*    wts   = (float*)(ws + WS_WTS);

  absmean_k    <<<3,    256, 0, stream>>>(gw, uw, dw, gamma);
  pack_expert_k<<<1536, 256, 0, stream>>>(gw, uw, dw, gamma, pg, pu, pd);
  pack_router_k<<<1056, 256, 0, stream>>>(w1, w2, p1, p2);
  router_k     <<<TTILES, 256, 0, stream>>>(x, p1, p2, wts);
  expert_k     <<<TTILES, 256, 0, stream>>>(x, pg, pu, pd, wts, out);
}
